// LSTM_82703890252422
// MI455X (gfx1250) — matrix operand order repaired and checked by reading, not compiled, not run on hardware
//
#include <hip/hip_runtime.h>
#include <hip/hip_bf16.h>

// ---------------------------------------------------------------------------
// LSTM step on MI455X (gfx1250, wave32, WMMA + TDM).
//   FLOPs ~371G, essential bytes ~650MB -> compute-bound; f16 WMMA 16x16x32
//   (f32 accum) gives 8x K/instr over f32 WMMA 16x16x4.
//   A-tiles staged by the Tensor Data Mover (tensor_load_to_lds) with HW
//   padding reproducing the 32+8-half LDS row layout; B-tiles staged through
//   registers (K-pair pack). Double-buffered LDS, one barrier per K-chunk,
//   TDM/next-loads overlap the 16 WMMAs of the current chunk.
// ---------------------------------------------------------------------------

typedef __attribute__((ext_vector_type(16))) _Float16 v16h;
typedef __attribute__((ext_vector_type(8)))  _Float16 v8h;
typedef __attribute__((ext_vector_type(8)))  float    v8f;
typedef __attribute__((ext_vector_type(4)))  float    v4f;
typedef __attribute__((ext_vector_type(4)))  unsigned int v4u;
typedef __attribute__((ext_vector_type(4)))  unsigned int u32x4;
typedef __attribute__((ext_vector_type(8)))  int          i32x8;
typedef __attribute__((ext_vector_type(4)))  int          i32x4;

#define WMMA16(a, b, c) \
  __builtin_amdgcn_wmma_f32_16x16x32_f16(false, (a), false, (b), (short)0, (c), false, false)

// ---- TDM descriptor helpers (D# layout per CDNA5 ISA ch. 7/8) -------------
// 2-D/3-D f16 tile load, global row-major -> LDS with 4-DWORD pad every
// 16 DWORDs (i.e. 32-half rows padded to 40 halves).
__device__ __forceinline__ void tdm_load_f16(
    unsigned lds_addr, const _Float16* gptr,
    unsigned tile_d0, unsigned tile_d1, unsigned tile_d2,
    unsigned long long tensor_d0, unsigned long long tensor_d1,
    unsigned long long tensor_d2,
    unsigned long long d0_stride, unsigned long long d1_stride) {
  unsigned long long ga = (unsigned long long)(size_t)gptr;
  u32x4 g0;
  g0[0] = 1u;                                        // count=1, user D#
  g0[1] = lds_addr;                                  // LDS byte address
  g0[2] = (unsigned)(ga & 0xFFFFFFFFu);              // global_addr[31:0]
  g0[3] = (unsigned)((ga >> 32) & 0x01FFFFFFu)       // global_addr[56:32]
          | (2u << 30);                              // type=2 ("image")
  i32x8 g1;
  // data_size=1 (2B) | pad_enable | pad_interval=3 (16 DW) | pad_amount=3 (4 DW)
  g1[0] = (int)((1u << 16) | (1u << 20) | (3u << 22) | (3u << 25));
  g1[1] = (int)((unsigned)(tensor_d0 & 0xFFFFu) << 16);          // dim0 lo16
  g1[2] = (int)(((unsigned)(tensor_d0 >> 16) & 0xFFFFu)          // dim0 hi16
                | ((unsigned)(tensor_d1 & 0xFFFFu) << 16));      // dim1 lo16
  g1[3] = (int)(((unsigned)(tensor_d1 >> 16) & 0xFFFFu)          // dim1 hi16
                | (tile_d0 << 16));                              // tile_dim0
  g1[4] = (int)((tile_d1 & 0xFFFFu) | (tile_d2 << 16));          // tile_dim1/2
  g1[5] = (int)(unsigned)(d0_stride & 0xFFFFFFFFu);              // d0 stride lo32
  g1[6] = (int)(((unsigned)((d0_stride >> 32) & 0xFFFFu))        // d0 stride hi16
                | ((unsigned)(d1_stride & 0xFFFFu) << 16));      // d1 stride lo16
  g1[7] = (int)(unsigned)((d1_stride >> 16) & 0xFFFFFFFFu);      // d1 stride [47:16]
  i32x4 g2;
  g2[0] = (int)(unsigned)tensor_d2;                  // tensor_dim2
  g2[1] = 0; g2[2] = 0; g2[3] = 0;
  i32x4 g3 = {0, 0, 0, 0};
  i32x8 g4 = {0, 0, 0, 0, 0, 0, 0, 0};
  __builtin_amdgcn_tensor_load_to_lds(g0, g1, g2, g3, g4, 0);
}

// --------------------------- f32 -> f16 convert ----------------------------
__global__ __launch_bounds__(256)
void k_f32_to_f16(const float* __restrict__ src, _Float16* __restrict__ dst,
                  long long n) {
  long long i = ((long long)blockIdx.x * 256 + threadIdx.x) * 8;
  if (i + 8 <= n) {
    v4f a = *(const v4f*)(src + i);
    v4f b = *(const v4f*)(src + i + 4);
    v8h o;
    o[0] = (_Float16)a[0]; o[1] = (_Float16)a[1];
    o[2] = (_Float16)a[2]; o[3] = (_Float16)a[3];
    o[4] = (_Float16)b[0]; o[5] = (_Float16)b[1];
    o[6] = (_Float16)b[2]; o[7] = (_Float16)b[3];
    *(v8h*)(dst + i) = o;
  }
}

// ------------------- embedding gather -> z16 lower rows --------------------
__global__ __launch_bounds__(256)
void k_gather_embed(const int* __restrict__ ids, const float* __restrict__ emb,
                    _Float16* __restrict__ z16) {
  int gid = blockIdx.x * 256 + threadIdx.x;  // B * (DIM/8) = 262144 threads
  int b  = gid >> 7;
  int dc = (gid & 127) << 3;
  long long row = ids[b];
  const float* src = emb + row * 1024 + dc;
  v4f a = *(const v4f*)src;
  v4f c = *(const v4f*)(src + 4);
  float vals[8] = {a[0], a[1], a[2], a[3], c[0], c[1], c[2], c[3]};
#pragma unroll
  for (int j = 0; j < 8; ++j)
    z16[(long long)(2048 + dc + j) * 2048 + b] = (_Float16)vals[j];
}

// --------------------- fused 4-gate GEMM + LSTM cell -----------------------
// W4 = [f;i;o;C] packed [4*2048, 3072] f16. Block = 64 n-rows x 64 b-cols for
// all four gates. A-tile (4 x 64 x 32) fetched by one 3-D TDM descriptor.
__global__ __launch_bounds__(256)
void k_gates(const _Float16* __restrict__ W4, const _Float16* __restrict__ Z,
             const float* __restrict__ pbf, const float* __restrict__ pbi,
             const float* __restrict__ pbo, const float* __restrict__ pbc,
             const float* __restrict__ Cprev,
             float* __restrict__ Cout, float* __restrict__ Hout,
             _Float16* __restrict__ H16) {
  __shared__ _Float16 As[2][4 * 64 * 40];    // 2 x 20,480 B
  __shared__ unsigned int Bs[2][16 * 68];    // 2 x  4,352 B
  __shared__ float Gs[4][64][64];            // 64 KB combine buffer

  const int tid  = threadIdx.x;
  const int lane = tid & 31;
  const int wave = tid >> 5;
  const int g    = wave >> 1;
  const int hm   = wave & 1;
  const int n0   = blockIdx.x * 64;
  const int b0   = blockIdx.y * 64;
  const int KZ = 3072, NB = 2048;

  v8f zero8 = {0.f, 0.f, 0.f, 0.f, 0.f, 0.f, 0.f, 0.f};
  v8f acc[2][4];
#pragma unroll
  for (int i = 0; i < 2; ++i)
#pragma unroll
    for (int j = 0; j < 4; ++j) acc[i][j] = zero8;

  // B staging (threads < 128): K-pair bkp, 8-col chunk
  const int bkp = (tid & 127) >> 3;
  const int bc  = (tid & 7) * 8;
  const bool doB = tid < 128;
  const _Float16* Bp = Z + (long long)(2 * bkp) * NB + b0 + bc;

  const int l15 = lane & 15;
  const int kb  = (lane >> 4) * 8;

  // A-tile TDM source: 3-D tensor x=K, y=row-in-gate, z=gate
  const _Float16* Abase = W4 + (long long)n0 * KZ;

  // ---- prologue: chunk 0 ----
  if (wave == 0)
    tdm_load_f16((unsigned)(size_t)&As[0][0], Abase,
                 /*tile*/ 32, 64, 4,
                 /*tensor*/ KZ, 2048, 4,
                 /*strides*/ KZ, 2048LL * KZ);
  v8h rbl = {}, rbh = {};
  if (doB) { rbl = *(const v8h*)Bp; rbh = *(const v8h*)(Bp + NB); }
  Bp += 32LL * NB;
  if (doB) {
    const unsigned short* lo = (const unsigned short*)&rbl;
    const unsigned short* hi = (const unsigned short*)&rbh;
    unsigned int p[8];
#pragma unroll
    for (int j = 0; j < 8; ++j)
      p[j] = (unsigned int)lo[j] | ((unsigned int)hi[j] << 16);
    v4u q0 = {p[0], p[1], p[2], p[3]};
    v4u q1 = {p[4], p[5], p[6], p[7]};
    *(v4u*)(&Bs[0][bkp * 68 + bc])     = q0;
    *(v4u*)(&Bs[0][bkp * 68 + bc + 4]) = q1;
  }
  if (wave == 0) __builtin_amdgcn_s_wait_tensorcnt(0);
  __syncthreads();

  int cur = 0;
  for (int k0 = 0; k0 < KZ; k0 += 32) {
    const int nxt = cur ^ 1;
    const bool more = (k0 + 32) < KZ;
    if (more) {
      if (wave == 0)
        tdm_load_f16((unsigned)(size_t)&As[nxt][0], Abase + (k0 + 32),
                     32, 64, 4, KZ, 2048, 4, KZ, 2048LL * KZ);
      if (doB) { rbl = *(const v8h*)Bp; rbh = *(const v8h*)(Bp + NB); }
      Bp += 32LL * NB;
    }

    const _Float16* asb = &As[cur][0];
    const unsigned int* bsb = &Bs[cur][0];
    v16h af[2];
#pragma unroll
    for (int mt = 0; mt < 2; ++mt) {
      int r = hm * 32 + mt * 16 + l15;
      const _Float16* base = asb + g * (64 * 40) + r * 40;
      v8h* ph = (v8h*)&af[mt];
      ph[0] = *(const v8h*)(base + kb);
      ph[1] = *(const v8h*)(base + kb + 16);
    }
#pragma unroll
    for (int nt = 0; nt < 4; ++nt) {
      int c = nt * 16 + l15;
      v16h bfv;
      unsigned int* bw = (unsigned int*)&bfv;
#pragma unroll
      for (int t = 0; t < 8; ++t) bw[t] = bsb[((kb >> 1) + (t & 3) + ((t >> 2) << 3)) * 68 + c];
#pragma unroll
      for (int mt = 0; mt < 2; ++mt)
        acc[mt][nt] = WMMA16(af[mt], bfv, acc[mt][nt]);
    }

    if (more) {
      if (doB) {
        const unsigned short* lo = (const unsigned short*)&rbl;
        const unsigned short* hi = (const unsigned short*)&rbh;
        unsigned int p[8];
#pragma unroll
        for (int j = 0; j < 8; ++j)
          p[j] = (unsigned int)lo[j] | ((unsigned int)hi[j] << 16);
        v4u q0 = {p[0], p[1], p[2], p[3]};
        v4u q1 = {p[4], p[5], p[6], p[7]};
        *(v4u*)(&Bs[nxt][bkp * 68 + bc])     = q0;
        *(v4u*)(&Bs[nxt][bkp * 68 + bc + 4]) = q1;
      }
      if (wave == 0) __builtin_amdgcn_s_wait_tensorcnt(0);
    }
    __syncthreads();
    cur = nxt;
  }

  // spill gate partials to LDS for cross-gate combine
  const int rsel = (lane >> 4) * 8;
#pragma unroll
  for (int mt = 0; mt < 2; ++mt)
#pragma unroll
    for (int nt = 0; nt < 4; ++nt)
#pragma unroll
      for (int v = 0; v < 8; ++v)
        Gs[g][hm * 32 + mt * 16 + rsel + v][nt * 16 + l15] = acc[mt][nt][v];
  __syncthreads();

  // elementwise LSTM cell: 4096 elems / 256 threads
#pragma unroll 4
  for (int j = 0; j < 16; ++j) {
    int e = tid + j * 256;
    int r = e >> 6, c = e & 63;
    int n = n0 + r, b = b0 + c;
    long long idx = (long long)n * 2048 + b;
    float xf = Gs[0][r][c] + pbf[n];
    float xi = Gs[1][r][c] + pbi[n];
    float xo = Gs[2][r][c] + pbo[n];
    float xc = Gs[3][r][c] + pbc[n];
    float fv = 1.f / (1.f + expf(-xf));
    float iv = 1.f / (1.f + expf(-xi));
    float ov = 1.f / (1.f + expf(-xo));
    float cv = tanhf(xc);
    float Cn = fv * Cprev[idx] + iv * cv;
    float hn = ov * tanhf(Cn);
    Cout[idx] = Cn;
    Hout[idx] = hn;
    H16[idx]  = (_Float16)hn;
  }
}

// ---------------------- generic f16 GEMM + bias (y) ------------------------
// C[Ms,Ns] = A[Ms,Ks] * B[Ks,Ns] + bias[Ms].  256x128 block, BK=32,
// 8 waves (4 row x 2 col), each wave 64x64 = 4x4 WMMA tiles. TDM A-staging.
__global__ __launch_bounds__(256)
void k_gemm_f16_bias(const _Float16* __restrict__ A,
                     const _Float16* __restrict__ Bm,
                     const float* __restrict__ bias, float* __restrict__ C,
                     int Ms, int Ns, int Ks) {
  __shared__ _Float16 As[2][256 * 40];       // 2 x 20,480 B
  __shared__ unsigned int Bs[2][16 * 132];   // 2 x  8,448 B

  const int tid  = threadIdx.x;
  const int lane = tid & 31;
  const int wave = tid >> 5;
  const int wm   = wave & 3;                 // 4 row-waves x 64 rows
  const int wn   = wave >> 2;                // 2 col-waves x 64 cols
  const int row0 = blockIdx.x * 256;
  const int col0 = blockIdx.y * 128;

  v8f zero8 = {0.f, 0.f, 0.f, 0.f, 0.f, 0.f, 0.f, 0.f};
  v8f acc[4][4];
#pragma unroll
  for (int i = 0; i < 4; ++i)
#pragma unroll
    for (int j = 0; j < 4; ++j) acc[i][j] = zero8;

  const int bkp = tid >> 4;
  const int bc  = (tid & 15) * 8;
  const _Float16* Bp = Bm + (long long)(2 * bkp) * Ns + col0 + bc;

  const int l15 = lane & 15;
  const int kb  = (lane >> 4) * 8;

  const _Float16* Abase = A + (long long)row0 * Ks;

  // ---- prologue: chunk 0 ----
  if (wave == 0)
    tdm_load_f16((unsigned)(size_t)&As[0][0], Abase,
                 /*tile*/ 32, 256, 0,
                 /*tensor*/ (unsigned long long)Ks, (unsigned long long)Ms, 0,
                 /*strides*/ (unsigned long long)Ks, 0);
  v8h rbl = *(const v8h*)(Bp);
  v8h rbh = *(const v8h*)(Bp + Ns);
  Bp += 32LL * Ns;
  {
    const unsigned short* lo = (const unsigned short*)&rbl;
    const unsigned short* hi = (const unsigned short*)&rbh;
    unsigned int p[8];
#pragma unroll
    for (int j = 0; j < 8; ++j)
      p[j] = (unsigned int)lo[j] | ((unsigned int)hi[j] << 16);
    v4u q0 = {p[0], p[1], p[2], p[3]};
    v4u q1 = {p[4], p[5], p[6], p[7]};
    *(v4u*)(&Bs[0][bkp * 132 + bc])     = q0;
    *(v4u*)(&Bs[0][bkp * 132 + bc + 4]) = q1;
  }
  if (wave == 0) __builtin_amdgcn_s_wait_tensorcnt(0);
  __syncthreads();

  int cur = 0;
  for (int k0 = 0; k0 < Ks; k0 += 32) {
    const int nxt = cur ^ 1;
    const bool more = (k0 + 32) < Ks;
    if (more) {
      if (wave == 0)
        tdm_load_f16((unsigned)(size_t)&As[nxt][0], Abase + (k0 + 32),
                     32, 256, 0, (unsigned long long)Ks,
                     (unsigned long long)Ms, 0, (unsigned long long)Ks, 0);
      rbl = *(const v8h*)(Bp);
      rbh = *(const v8h*)(Bp + Ns);
      Bp += 32LL * Ns;
      __builtin_prefetch(Bp, 0, 3);
    }

    const _Float16* asb = &As[cur][0];
    const unsigned int* bsb = &Bs[cur][0];
    v16h af[4];
#pragma unroll
    for (int mt = 0; mt < 4; ++mt) {
      int r = wm * 64 + mt * 16 + l15;
      v8h* ph = (v8h*)&af[mt];
      ph[0] = *(const v8h*)(asb + r * 40 + kb);
      ph[1] = *(const v8h*)(asb + r * 40 + kb + 16);
    }
#pragma unroll
    for (int nt = 0; nt < 4; ++nt) {
      int c = wn * 64 + nt * 16 + l15;
      v16h bfv;
      unsigned int* bw = (unsigned int*)&bfv;
#pragma unroll
      for (int t = 0; t < 8; ++t) bw[t] = bsb[((kb >> 1) + (t & 3) + ((t >> 2) << 3)) * 132 + c];
#pragma unroll
      for (int mt = 0; mt < 4; ++mt)
        acc[mt][nt] = WMMA16(af[mt], bfv, acc[mt][nt]);
    }

    if (more) {
      const unsigned short* lo = (const unsigned short*)&rbl;
      const unsigned short* hi = (const unsigned short*)&rbh;
      unsigned int p[8];
#pragma unroll
      for (int j = 0; j < 8; ++j)
        p[j] = (unsigned int)lo[j] | ((unsigned int)hi[j] << 16);
      v4u q0 = {p[0], p[1], p[2], p[3]};
      v4u q1 = {p[4], p[5], p[6], p[7]};
      *(v4u*)(&Bs[nxt][bkp * 132 + bc])     = q0;
      *(v4u*)(&Bs[nxt][bkp * 132 + bc + 4]) = q1;
      if (wave == 0) __builtin_amdgcn_s_wait_tensorcnt(0);
    }
    __syncthreads();
    cur = nxt;
  }

  const int rsel = (lane >> 4) * 8;
#pragma unroll
  for (int mt = 0; mt < 4; ++mt) {
    int rbase = row0 + wm * 64 + mt * 16 + rsel;
#pragma unroll
    for (int nt = 0; nt < 4; ++nt) {
      int c = col0 + wn * 64 + nt * 16 + l15;
#pragma unroll
      for (int v = 0; v < 8; ++v)
        C[(long long)(rbase + v) * Ns + c] = acc[mt][nt][v] + bias[rbase + v];
    }
  }
}

// ---------------------------------------------------------------------------
extern "C" void kernel_launch(void* const* d_in, const int* in_sizes, int n_in,
                              void* d_out, int out_size, void* d_ws,
                              size_t ws_size, hipStream_t stream) {
  (void)in_sizes; (void)n_in; (void)out_size; (void)ws_size;
  const int M = 32000, N = 2048, DIM = 1024, B = 2048;
  const int KZ = N + DIM;  // 3072

  const int*   ids    = (const int*)d_in[0];
  const float* h_prev = (const float*)d_in[1];
  const float* C_prev = (const float*)d_in[2];
  const float* emb    = (const float*)d_in[3];
  const float* W_A    = (const float*)d_in[4];
  const float* W_C    = (const float*)d_in[5];
  const float* W_f    = (const float*)d_in[6];
  const float* W_o    = (const float*)d_in[7];
  const float* W_i    = (const float*)d_in[8];
  const float* b_A    = (const float*)d_in[9];
  const float* b_C    = (const float*)d_in[10];
  const float* b_f    = (const float*)d_in[11];
  const float* b_o    = (const float*)d_in[12];
  const float* b_i    = (const float*)d_in[13];

  char* ws = (char*)d_ws;
  _Float16* z16  = (_Float16*)(ws);               // [3072,2048]  12.6 MB
  _Float16* W4   = (_Float16*)(ws + 12582912);    // [8192,3072]  50.3 MB
  _Float16* WA16 = (_Float16*)(ws + 62914560);    // [32000,2048] 131 MB
  _Float16* H16  = (_Float16*)(ws + 193986560);   // [2048,2048]  8.4 MB

  float* y_out = (float*)d_out;
  float* h_out = y_out + (long long)M * B;
  float* C_out = h_out + (long long)N * B;

  long long nh = (long long)N * B;
  k_f32_to_f16<<<(unsigned)((nh / 8 + 255) / 256), 256, 0, stream>>>(h_prev, z16, nh);
  k_gather_embed<<<(B * (DIM / 8)) / 256, 256, 0, stream>>>(ids, emb, z16);
  long long gsz = (long long)N * KZ;
  unsigned gblk = (unsigned)((gsz / 8 + 255) / 256);
  k_f32_to_f16<<<gblk, 256, 0, stream>>>(W_f, W4 + 0 * gsz, gsz);
  k_f32_to_f16<<<gblk, 256, 0, stream>>>(W_i, W4 + 1 * gsz, gsz);
  k_f32_to_f16<<<gblk, 256, 0, stream>>>(W_o, W4 + 2 * gsz, gsz);
  k_f32_to_f16<<<gblk, 256, 0, stream>>>(W_C, W4 + 3 * gsz, gsz);
  long long na = (long long)M * N;
  k_f32_to_f16<<<(unsigned)((na / 8 + 255) / 256), 256, 0, stream>>>(W_A, WA16, na);

  k_gates<<<dim3(N / 64, B / 64), 256, 0, stream>>>(
      W4, z16, b_f, b_i, b_o, b_C, C_prev, C_out, h_out, H16);

  k_gemm_f16_bias<<<dim3(M / 256, B / 128), 256, 0, stream>>>(
      WA16, H16, b_A, y_out, M, B, N);
}
